// GAT_55791625175364
// MI455X (gfx1250) — matrix operand order repaired and checked by reading, not compiled, not run on hardware
//
#include <hip/hip_runtime.h>
#include <math.h>

#define NHEAD 4
#define FEAT  64
#define HF    256   // NHEAD*FEAT
#define NEG_SLOPE 0.2f

typedef __attribute__((ext_vector_type(16))) __bf16 v16bf;
typedef __attribute__((ext_vector_type(8)))  float  v8f;
typedef __attribute__((ext_vector_type(4)))  unsigned int u32x4;
typedef __attribute__((ext_vector_type(4)))  int    i32x4;
typedef __attribute__((ext_vector_type(8)))  int    i32x8;

// ---------- helpers ----------
__device__ __forceinline__ __bf16 f2bf(float f) {
    union { float f; unsigned u; } x; x.f = f;
    unsigned r = (x.u + 0x7FFFu + ((x.u >> 16) & 1u)) >> 16;   // round-to-nearest-even
    union { unsigned short s; __bf16 b; } y; y.s = (unsigned short)r;
    return y.b;
}

__device__ __forceinline__ void atomicMaxFloat(float* addr, float v) {
    // signed/unsigned bit trick; valid with -inf initialization
    if (v >= 0.0f) atomicMax((int*)addr, __float_as_int(v));
    else           atomicMin((unsigned int*)addr, __float_as_uint(v));
}

// TDM: async 2-D tile load (bf16 elements) from global into LDS.
// D# group0: count=1 | lds_addr | global_addr | type=2
// D# group1: data_size=2B, tensor_dim0=cols, tensor_dim1=rowsLeft (zero-fill OOB),
//            tile_dim0=cols, tile_dim1=rows, dim0_stride=strideElems
__device__ __forceinline__ void tdm_load_tile_bf16(void* ldsDst, const __bf16* gsrc,
                                                   int tileCols, int tileRows,
                                                   int tensorRowsLeft, int strideElems) {
    unsigned long long ga = (unsigned long long)(size_t)gsrc;
    u32x4 g0;
    g0[0] = 1u;                                    // count=1 (valid), user mode
    g0[1] = (unsigned)(size_t)ldsDst;              // lds_addr (LDS byte offset)
    g0[2] = (unsigned)ga;                          // global_addr[31:0]
    g0[3] = (unsigned)((ga >> 32) & 0x01FFFFFFull) | (2u << 30);  // addr[56:32] | type=2
    i32x8 g1;
    g1[0] = (int)(1u << 16);                       // workgroup_mask=0, data_size=1 (2 bytes)
    g1[1] = (int)(((unsigned)tileCols & 0xFFFFu) << 16);                  // tensor_dim0 lo
    g1[2] = (int)(((unsigned)tileCols >> 16) |
                  (((unsigned)tensorRowsLeft & 0xFFFFu) << 16));          // td0 hi | td1 lo
    g1[3] = (int)(((unsigned)tensorRowsLeft >> 16) |
                  (((unsigned)tileCols & 0xFFFFu) << 16));                // td1 hi | tile_dim0
    g1[4] = (int)((unsigned)tileRows & 0xFFFFu);   // tile_dim1 | tile_dim2=0
    g1[5] = strideElems;                           // tensor_dim0_stride lo
    g1[6] = 0;
    g1[7] = 0;
    i32x4 z4 = {0, 0, 0, 0};
#if defined(__clang_major__) && (__clang_major__ >= 23)
    i32x8 z8 = {0, 0, 0, 0, 0, 0, 0, 0};
    __builtin_amdgcn_tensor_load_to_lds(g0, g1, z4, z4, z8, 0);
#else
    __builtin_amdgcn_tensor_load_to_lds(g0, g1, z4, z4, 0);
#endif
}

// ---------- fill ----------
__global__ void k_fill(float* __restrict__ p, float v, int n) {
    int i = blockIdx.x * blockDim.x + threadIdx.x;
    if (i < n) p[i] = v;
}

// ---------- f32 -> bf16 convert ----------
__global__ void k_f32_to_bf16(const float* __restrict__ s, __bf16* __restrict__ d, int n) {
    int i = blockIdx.x * blockDim.x + threadIdx.x;
    if (i < n) d[i] = f2bf(s[i]);
}

// ---------- weight: f32 [K x Nc] -> bf16 transposed [Nc x K] ----------
__global__ void k_w_to_bf16t(const float* __restrict__ W, __bf16* __restrict__ Bt, int K, int Nc) {
    int t = blockIdx.x * blockDim.x + threadIdx.x;
    if (t >= K * Nc) return;
    const int n = t / K, k = t - n * K;
    Bt[t] = f2bf(W[(size_t)k * Nc + n]);
}

// ---------- WMMA GEMM: C[M x Nc] = A[M x K](bf16) * Bt[Nc x K](bf16, pre-transposed) ----------
// A strip (16 rows x K) staged in LDS by the Tensor Data Mover.
__global__ __launch_bounds__(256)
void k_gemm_bf16(const __bf16* __restrict__ Abf, const __bf16* __restrict__ Bt,
                 float* __restrict__ C, int M, int K, int Nc) {
    extern __shared__ char smem_raw[];
    __bf16* shA = (__bf16*)smem_raw;                  // 16 x K bf16
    const int lane = threadIdx.x & 31;
    const int wave = threadIdx.x >> 5;                // 0..7
    const int m0   = blockIdx.x << 4;

    if (wave == 0) {
        tdm_load_tile_bf16(shA, Abf + (size_t)m0 * K, K, 16, M - m0, K);
        __builtin_amdgcn_s_wait_tensorcnt(0);
    }
    __syncthreads();

    const int rA = lane & 15;
    const int nTiles = Nc >> 4;
    for (int nt = wave; nt < nTiles; nt += 8) {
        const int n0 = nt << 4;
        const int colB = n0 + (lane & 15);
        const __bf16* bp = Bt + (size_t)colB * K;
        v8f acc = {};
        for (int k0 = 0; k0 < K; k0 += 32) {
            // A fragment 16x32: lane row=lane%16; e<8 -> k0+e (+8 if lane>=16),
            //                   e>=8 -> +16 further. Two contiguous 16B LDS loads.
            union { uint4 q[2]; v16bf v; } fa, fb;
            const int kA = k0 + ((lane >= 16) ? 8 : 0);
            const __bf16* ap = shA + rA * K + kA;
            fa.q[0] = *(const uint4*)(ap);
            fa.q[1] = *(const uint4*)(ap + 16);
            // B fragment 32x16: lane col=lane%16; e<8 -> k0+e (+8 if lane>=16),
            //                   e>=8 -> +16 further. Two 16B global loads.
            const int kB = k0 + ((lane >= 16) ? 8 : 0);
            fb.q[0] = *(const uint4*)(bp + kB);
            fb.q[1] = *(const uint4*)(bp + kB + 16);
            acc = __builtin_amdgcn_wmma_f32_16x16x32_bf16(
                      false, fa.v, false, fb.v, (short)0, acc, false, false);
        }
        // C 16x16 f32: col = n0+lane%16, VGPR r -> row m0 + r + (lane>=16 ? 8 : 0)
        const int rbase = m0 + ((lane >= 16) ? 8 : 0);
        #pragma unroll
        for (int r = 0; r < 8; ++r) {
            const int row = rbase + r;
            if (row < M) C[(size_t)row * (size_t)Nc + colB] = acc[r];
        }
    }
}

// ---------- el/er: [N,H] attention projections ----------
__global__ void k_attn_coeff(const float* __restrict__ feat,
                             const float* __restrict__ al, const float* __restrict__ ar,
                             float* __restrict__ el, float* __restrict__ er, int Nn) {
    int t = blockIdx.x * blockDim.x + threadIdx.x;
    if (t >= Nn * NHEAD) return;
    const int n = t >> 2, h = t & 3;
    const float* fp  = feat + (size_t)n * HF + h * FEAT;
    const float* alp = al + h * FEAT;
    const float* arp = ar + h * FEAT;
    float sl = 0.f, sr = 0.f;
    #pragma unroll 8
    for (int f = 0; f < FEAT; ++f) { const float v = fp[f]; sl += v * alp[f]; sr += v * arp[f]; }
    el[t] = sl; er[t] = sr;
}

// ---------- edge pass 1: segment max of leaky_relu(el[src]+er[dst]) ----------
__global__ void k_edge_max(const int* __restrict__ src, const int* __restrict__ dst,
                           const float* __restrict__ el, const float* __restrict__ er,
                           float* __restrict__ m, int Ecnt) {
    int t = blockIdx.x * blockDim.x + threadIdx.x;
    if (t >= Ecnt * NHEAD) return;
    const int e = t >> 2, h = t & 3;
    const int s = src[e], d = dst[e];
    float v = el[s * NHEAD + h] + er[d * NHEAD + h];
    v = (v > 0.f) ? v : NEG_SLOPE * v;
    atomicMaxFloat(&m[d * NHEAD + h], v);
}

// ---------- edge pass 2: a = exp(v - m[dst]); denom += a; accum += a * feat[src] ----------
__global__ void k_edge_acc(const int* __restrict__ src, const int* __restrict__ dst,
                           const float* __restrict__ el, const float* __restrict__ er,
                           const float* __restrict__ m, const float* __restrict__ feat,
                           float* __restrict__ denom, float* __restrict__ accum, int Ecnt) {
    int t = blockIdx.x * blockDim.x + threadIdx.x;
    if (t >= Ecnt * NHEAD) return;
    const int e = t >> 2, h = t & 3;
    const int s = src[e], d = dst[e];
    float v = el[s * NHEAD + h] + er[d * NHEAD + h];
    v = (v > 0.f) ? v : NEG_SLOPE * v;
    const float a = __expf(v - m[d * NHEAD + h]);
    atomicAdd(&denom[d * NHEAD + h], a);
    const float4* fs = (const float4*)(feat + (size_t)s * HF + h * FEAT);
    float*        ac = accum + (size_t)d * HF + h * FEAT;
    #pragma unroll 4
    for (int q = 0; q < FEAT / 4; ++q) {
        const float4 v4 = fs[q];
        atomicAdd(&ac[4 * q + 0], a * v4.x);
        atomicAdd(&ac[4 * q + 1], a * v4.y);
        atomicAdd(&ac[4 * q + 2], a * v4.z);
        atomicAdd(&ac[4 * q + 3], a * v4.w);
    }
}

// ---------- finalize, flatten heads (layer 1, in-place on accum) ----------
__global__ void k_final_flat(float* __restrict__ accum, const float* __restrict__ denom,
                             const float* __restrict__ res, const float* __restrict__ bias, int Nn) {
    int t = blockIdx.x * blockDim.x + threadIdx.x;
    if (t >= Nn * HF) return;
    const int n = t >> 8, c = t & 255, h = c >> 6;
    float v = accum[t] / denom[n * NHEAD + h] + res[t] + bias[c];
    accum[t] = (v > 0.f) ? v : (__expf(v) - 1.f);     // ELU
}

// ---------- finalize, mean over heads (layer 2) ----------
__global__ void k_final_mean(const float* __restrict__ accum, const float* __restrict__ denom,
                             const float* __restrict__ res, const float* __restrict__ bias,
                             float* __restrict__ out, int Nn) {
    int t = blockIdx.x * blockDim.x + threadIdx.x;
    if (t >= Nn * FEAT) return;
    const int n = t >> 6, f = t & 63;
    float s = 0.f;
    #pragma unroll
    for (int h = 0; h < NHEAD; ++h) {
        const int c = h * FEAT + f;
        const size_t idx = (size_t)n * HF + c;
        float v = accum[idx] / denom[n * NHEAD + h] + res[idx] + bias[c];
        s += (v > 0.f) ? v : (__expf(v) - 1.f);
    }
    out[t] = s * 0.25f;
}

// ---------- per-node sigmoid gate ----------
__global__ void k_nodew(const float* __restrict__ h2, const float* __restrict__ Ww,
                        const float* __restrict__ bw, float* __restrict__ w, int Nn) {
    int n = blockIdx.x * blockDim.x + threadIdx.x;
    if (n >= Nn) return;
    float s = bw[0];
    #pragma unroll 8
    for (int f = 0; f < FEAT; ++f) s += h2[(size_t)n * FEAT + f] * Ww[f];
    w[n] = 1.f / (1.f + __expf(-s));
}

// ---------- weighted-sum + max readout per graph ----------
__global__ void k_readout(const float* __restrict__ h2, const float* __restrict__ w,
                          const int* __restrict__ gid, float* __restrict__ wsum,
                          float* __restrict__ hmax, int Nn) {
    int t = blockIdx.x * blockDim.x + threadIdx.x;
    if (t >= Nn * FEAT) return;
    const int n = t >> 6, f = t & 63;
    const int g = gid[n];
    const float v = h2[t];
    atomicAdd(&wsum[g * FEAT + f], v * w[n]);
    atomicMaxFloat(&hmax[g * FEAT + f], v);
}

// ---------- predictor MLP: [2F] -> 32 -> 1 per graph ----------
__global__ void k_pred(const float* __restrict__ wsum, const float* __restrict__ hmax,
                       const float* __restrict__ Wp1, const float* __restrict__ bp1,
                       const float* __restrict__ Wp2, const float* __restrict__ bp2,
                       float* __restrict__ out, int Gn) {
    int g = blockIdx.x * blockDim.x + threadIdx.x;
    if (g >= Gn) return;
    const float* ws = wsum + (size_t)g * FEAT;
    const float* hm = hmax + (size_t)g * FEAT;
    float acc = bp2[0];
    for (int p = 0; p < 32; ++p) {
        float s = bp1[p];
        #pragma unroll 8
        for (int i = 0; i < FEAT; ++i) s += ws[i] * Wp1[i * 32 + p];
        #pragma unroll 8
        for (int i = 0; i < FEAT; ++i) s += hm[i] * Wp1[(FEAT + i) * 32 + p];
        if (s < 0.f) s = 0.f;
        acc += s * Wp2[p];
    }
    out[g] = acc;
}

// ---------- launcher ----------
static inline int cdiv(int a, int b) { return (a + b - 1) / b; }

extern "C" void kernel_launch(void* const* d_in, const int* in_sizes, int n_in,
                              void* d_out, int out_size, void* d_ws, size_t ws_size,
                              hipStream_t stream) {
    const float* x     = (const float*)d_in[0];
    const int*   src   = (const int*)  d_in[1];
    const int*   dst   = (const int*)  d_in[2];
    const int*   gid   = (const int*)  d_in[3];
    const float* W1    = (const float*)d_in[4];
    const float* al1   = (const float*)d_in[5];
    const float* ar1   = (const float*)d_in[6];
    const float* b1    = (const float*)d_in[7];
    const float* Wres1 = (const float*)d_in[8];
    const float* W2    = (const float*)d_in[9];
    const float* al2   = (const float*)d_in[10];
    const float* ar2   = (const float*)d_in[11];
    const float* b2    = (const float*)d_in[12];
    const float* Wres2 = (const float*)d_in[13];
    const float* Ww    = (const float*)d_in[14];
    const float* bw    = (const float*)d_in[15];
    const float* Wp1   = (const float*)d_in[16];
    const float* bp1   = (const float*)d_in[17];
    const float* Wp2   = (const float*)d_in[18];
    const float* bp2   = (const float*)d_in[19];
    float* out = (float*)d_out;

    const int Nn   = in_sizes[0] / 64;   // nodes (IN = 64)
    const int Ecnt = in_sizes[1];        // edges
    const int Gn   = out_size;           // graphs

    // workspace carving (256B aligned)
    char* ws = (char*)d_ws;
    size_t off = 0;
    auto carve = [&](size_t bytes) -> char* {
        char* p = ws + off;
        off = (off + bytes + 255) & ~(size_t)255;
        return p;
    };
    float*  bufA  = (float*) carve((size_t)Nn * HF * 4);     // feat (both layers)
    float*  bufB  = (float*) carve((size_t)Nn * HF * 4);     // residual (both layers)
    float*  bufC  = (float*) carve((size_t)Nn * HF * 4);     // accum / h1
    float*  h2    = (float*) carve((size_t)Nn * FEAT * 4);   // layer-2 output
    __bf16* Abf   = (__bf16*)carve((size_t)Nn * HF * 2);     // bf16 GEMM input
    __bf16* Btb   = (__bf16*)carve((size_t)HF * HF * 2);     // bf16 transposed weights
    float*  el    = (float*) carve((size_t)Nn * NHEAD * 4);
    float*  er    = (float*) carve((size_t)Nn * NHEAD * 4);
    float*  mmax  = (float*) carve((size_t)Nn * NHEAD * 4);
    float*  denom = (float*) carve((size_t)Nn * NHEAD * 4);
    float*  nodew = (float*) carve((size_t)Nn * 4);
    float*  wsum  = (float*) carve((size_t)Gn * FEAT * 4);
    float*  hmax  = (float*) carve((size_t)Gn * FEAT * 4);
    (void)ws_size; (void)n_in;

    const float NEG_INF = -__builtin_huge_valf();
    const int TB = 256;
    const int gM   = cdiv(Nn, 16);           // GEMM M-strips
    const int gNH  = cdiv(Nn * NHEAD, TB);
    const int gEH  = cdiv(Ecnt * NHEAD, TB);
    const int gNHF = cdiv(Nn * HF, TB);
    const int gNF  = cdiv(Nn * FEAT, TB);
    const size_t shm1 = 16 * 64 * sizeof(__bf16);   // K=64 strip
    const size_t shm2 = 16 * HF * sizeof(__bf16);   // K=256 strip

    // ===== layer 1 (flatten) =====
    k_f32_to_bf16<<<cdiv(Nn * 64, TB), TB, 0, stream>>>(x, Abf, Nn * 64);
    k_w_to_bf16t<<<cdiv(64 * HF, TB), TB, 0, stream>>>(W1, Btb, 64, HF);
    k_gemm_bf16<<<gM, TB, shm1, stream>>>(Abf, Btb, bufA, Nn, 64, HF);     // feat1
    k_w_to_bf16t<<<cdiv(64 * HF, TB), TB, 0, stream>>>(Wres1, Btb, 64, HF);
    k_gemm_bf16<<<gM, TB, shm1, stream>>>(Abf, Btb, bufB, Nn, 64, HF);     // res1
    k_fill<<<gNH,  TB, 0, stream>>>(mmax,  NEG_INF, Nn * NHEAD);
    k_fill<<<gNH,  TB, 0, stream>>>(denom, 0.f,     Nn * NHEAD);
    k_fill<<<gNHF, TB, 0, stream>>>(bufC,  0.f,     Nn * HF);
    k_attn_coeff<<<gNH, TB, 0, stream>>>(bufA, al1, ar1, el, er, Nn);
    k_edge_max<<<gEH, TB, 0, stream>>>(src, dst, el, er, mmax, Ecnt);
    k_edge_acc<<<gEH, TB, 0, stream>>>(src, dst, el, er, mmax, bufA, denom, bufC, Ecnt);
    k_final_flat<<<gNHF, TB, 0, stream>>>(bufC, denom, bufB, b1, Nn);      // h1 in bufC

    // ===== layer 2 (mean) =====
    k_f32_to_bf16<<<gNHF, TB, 0, stream>>>(bufC, Abf, Nn * HF);
    k_w_to_bf16t<<<cdiv(HF * HF, TB), TB, 0, stream>>>(W2, Btb, HF, HF);
    k_gemm_bf16<<<gM, TB, shm2, stream>>>(Abf, Btb, bufA, Nn, HF, HF);     // feat2
    k_w_to_bf16t<<<cdiv(HF * HF, TB), TB, 0, stream>>>(Wres2, Btb, HF, HF);
    k_gemm_bf16<<<gM, TB, shm2, stream>>>(Abf, Btb, bufB, Nn, HF, HF);     // res2
    k_fill<<<gNH,  TB, 0, stream>>>(mmax,  NEG_INF, Nn * NHEAD);
    k_fill<<<gNH,  TB, 0, stream>>>(denom, 0.f,     Nn * NHEAD);
    k_fill<<<gNHF, TB, 0, stream>>>(bufC,  0.f,     Nn * HF);              // accum2 reuses bufC
    k_attn_coeff<<<gNH, TB, 0, stream>>>(bufA, al2, ar2, el, er, Nn);
    k_edge_max<<<gEH, TB, 0, stream>>>(src, dst, el, er, mmax, Ecnt);
    k_edge_acc<<<gEH, TB, 0, stream>>>(src, dst, el, er, mmax, bufA, denom, bufC, Ecnt);
    k_final_mean<<<gNF, TB, 0, stream>>>(bufC, denom, bufB, b2, h2, Nn);

    // ===== readout + predictor =====
    k_fill<<<cdiv(Gn * FEAT, TB), TB, 0, stream>>>(wsum, 0.f,     Gn * FEAT);
    k_fill<<<cdiv(Gn * FEAT, TB), TB, 0, stream>>>(hmax, NEG_INF, Gn * FEAT);
    k_nodew<<<cdiv(Nn, TB), TB, 0, stream>>>(h2, Ww, bw, nodew, Nn);
    k_readout<<<gNF, TB, 0, stream>>>(h2, nodew, gid, wsum, hmax, Nn);
    k_pred<<<cdiv(Gn, TB), TB, 0, stream>>>(wsum, hmax, Wp1, bp1, Wp2, bp2, out, Gn);
}
